// DomainEmbedding_10496900071806
// MI455X (gfx1250) — matrix operand order repaired and checked by reading, not compiled, not run on hardware
//
#include <hip/hip_runtime.h>
#include <cmath>

typedef __attribute__((ext_vector_type(16))) _Float16 v16h;
typedef __attribute__((ext_vector_type(8)))  _Float16 v8h;
typedef __attribute__((ext_vector_type(8)))  float    v8f;
typedef __attribute__((ext_vector_type(4)))  float    v4f;

#define VOCAB 50257
#define E  64
#define D  16
#define DK 32
#define NTOK (32 * 2048)          // B*S = 65536 tokens
#define NTILES (NTOK / 16)        // 4096 wave-tiles of 16 tokens
#define WAVES_PER_WG 8
#define NUM_WG 128

// Branchless GELU: x * sigmoid(1.702 x).
// v_mul + v_exp_f32 + v_add + v_rcp_f32 + v_mul -- no division sequence, no divergence.
__device__ __forceinline__ float gelu_fast(float x) {
    float e = __expf(-1.702f * x);
    float s = __builtin_amdgcn_rcpf(1.0f + e);
    return x * s;
}

__global__ __launch_bounds__(256) void domain_embed_wmma(
    const int* __restrict__ x,              // [B*S] token ids
    const float* __restrict__ embed,        // [VOCAB][E]
    const float* __restrict__ W1,           // [D][E][DK]
    const float* __restrict__ W2,           // [D][DK][E]
    const unsigned char* __restrict__ memb, // [VOCAB][D] bool bytes
    float* __restrict__ out)                // [B*S][E]
{
    // Weights staged once per workgroup, transposed, f16, in B-operand-friendly order.
    __shared__ _Float16 lW1t[D * DK * E];             // [d][dk][e]  (64 KB)
    __shared__ _Float16 lW2t[D * E * DK];             // [d][e][dk]  (64 KB)
    __shared__ _Float16 lU[WAVES_PER_WG][16][DK];     // per-wave u staging (8 KB)

    const int tid = threadIdx.x;

    // ---- Stage W1 transposed: lW1t[d][dk][e] = W1[d][e][dk] ----
    for (int i = tid; i < D * E * DK; i += blockDim.x) {
        int d = i / (E * DK);
        int r = i % (E * DK);
        int e = r / DK;
        int k = r % DK;
        lW1t[(d * DK + k) * E + e] = (_Float16)W1[i];
    }
    // ---- Stage W2 transposed: lW2t[d][e][dk] = W2[d][dk][e] ----
    for (int i = tid; i < D * DK * E; i += blockDim.x) {
        int d = i / (DK * E);
        int r = i % (DK * E);
        int k = r / E;
        int e = r % E;
        lW2t[(d * E + e) * DK + k] = (_Float16)W2[i];
    }
    __syncthreads();

    const int wv   = tid >> 5;       // wave in workgroup (wave32)
    const int lane = tid & 31;
    const int hi   = lane >> 4;      // lane half: 0 or 1
    const int lm   = lane & 15;      // row (A) / column (B,C,D) index

    const int waveGlobal = blockIdx.x * WAVES_PER_WG + wv;
    const int nWaves     = gridDim.x * WAVES_PER_WG;

    for (int tile = waveGlobal; tile < NTILES; tile += nWaves) {
        const int tokBase = tile * 16;
        const int myTok   = x[tokBase + lm];   // token for this lane's A-row

        // ---- Membership: one 16B row load per lane, packed to a 16-bit mask ----
        // myBits bit d == token(lane row lm) belongs to domain d.
        unsigned myBits = 0;
        {
            const uint4 mrow = *(const uint4*)(memb + (long)myTok * D); // 16B aligned
            unsigned w0 = mrow.x, w1 = mrow.y, w2 = mrow.z, w3 = mrow.w;
            #pragma unroll
            for (int b = 0; b < 4; ++b) {
                myBits |= ((w0 >> (8 * b)) & 1u) << (b);
                myBits |= ((w1 >> (8 * b)) & 1u) << (4 + b);
                myBits |= ((w2 >> (8 * b)) & 1u) << (8 + b);
                myBits |= ((w3 >> (8 * b)) & 1u) << (12 + b);
            }
        }
        // Wave-OR reduce -> which domains have ANY member among these 16 tokens.
        unsigned waveBits = myBits;
        #pragma unroll
        for (int off = 16; off >= 1; off >>= 1)
            waveBits |= (unsigned)__shfl_xor((int)waveBits, off, 32);
        const unsigned domMask = (unsigned)__builtin_amdgcn_readfirstlane((int)waveBits);

        // ---- Build A1: 16x64 f16 embedding tile as two 16x32 A operands ----
        // A k order used here (B matches): lane(lm,hi): elem j -> k = 16*hi + j
        v16h a1[2];
        {
            const float* hrow = embed + (long)myTok * E;
            #pragma unroll
            for (int kc = 0; kc < 2; ++kc) {
                v4f f0 = *(const v4f*)(hrow + 32 * kc + 16 * hi);
                v4f f1 = *(const v4f*)(hrow + 32 * kc + 16 * hi + 4);
                v4f f2 = *(const v4f*)(hrow + 32 * kc + 16 * hi + 8);
                v4f f3 = *(const v4f*)(hrow + 32 * kc + 16 * hi + 12);
                v16h a;
                #pragma unroll
                for (int j = 0; j < 4; ++j) {
                    a[j]      = (_Float16)f0[j];
                    a[4 + j]  = (_Float16)f1[j];
                    a[8 + j]  = (_Float16)f2[j];
                    a[12 + j] = (_Float16)f3[j];
                }
                a1[kc] = a;
            }
        }

        v8f accOut[4] = {v8f{}, v8f{}, v8f{}, v8f{}};   // 4 N-tiles over E=64

        for (int d = 0; d < D; ++d) {
            // Scalar (SGPR) skip: EXEC untouched, no memory ops.
            if (!((domMask >> d) & 1u)) continue;

            // ---- 1st GEMM: u(16xDK) = h(16x64) @ W1[d](64xDK) ----
            v8f acc0 = {}, acc1 = {};
            #pragma unroll
            for (int kc = 0; kc < 2; ++kc) {
                // B elem j = B[16*hi + j][lm] = W1[d][e = 32*kc + 16*hi + j][dk = nt*16 + lm]
                const _Float16* b0p = &lW1t[(d * DK + 0 * 16 + lm) * E + 32 * kc + 16 * hi];
                const _Float16* b1p = &lW1t[(d * DK + 1 * 16 + lm) * E + 32 * kc + 16 * hi];
                v16h b0 = *(const v16h*)b0p;
                v16h b1 = *(const v16h*)b1p;
                acc0 = __builtin_amdgcn_wmma_f32_16x16x32_f16(false, a1[kc], false, b0,
                                                              (short)0, acc0, false, false);
                acc1 = __builtin_amdgcn_wmma_f32_16x16x32_f16(false, a1[kc], false, b1,
                                                              (short)0, acc1, false, false);
            }

            // ---- branchless GELU, stage u (D-layout -> LDS [m][k] row-major) ----
            // C/D layout: VGPR r holds m = r + 8*hi, n = lm
            #pragma unroll
            for (int r = 0; r < 8; ++r) {
                int m = r + 8 * hi;
                lU[wv][m][lm]      = (_Float16)gelu_fast(acc0[r]);  // dk = lm
                lU[wv][m][16 + lm] = (_Float16)gelu_fast(acc1[r]);  // dk = 16+lm
            }
            // Cross-lane restage within the wave: fence LDS stores before loads.
            asm volatile("s_wait_dscnt 0" ::: "memory");

            // ---- A2: u as 16x32 A operand, row m = lm, scaled by 0.1*mask ----
            v8h ulo = *(const v8h*)&lU[wv][lm][16 * hi];
            v8h uhi = *(const v8h*)&lU[wv][lm][16 * hi + 8];
            _Float16 sm = (_Float16)(0.1f * (float)((myBits >> d) & 1u));
            v16h a2;
            #pragma unroll
            for (int j = 0; j < 8; ++j) {
                a2[j]     = ulo[j] * sm;
                a2[8 + j] = uhi[j] * sm;
            }

            // ---- 2nd GEMM: c(16xE) += u(16x32) @ W2[d](32xE), chained over d ----
            #pragma unroll
            for (int nt = 0; nt < 4; ++nt) {
                // B elem j = W2[d][k = 16*hi + j][e = nt*16 + lm] = lW2t[d][e][k]
                const _Float16* b2p = &lW2t[(d * E + nt * 16 + lm) * DK + 16 * hi];
                v16h b2 = *(const v16h*)b2p;
                accOut[nt] = __builtin_amdgcn_wmma_f32_16x16x32_f16(false, a2, false, b2,
                                                                    (short)0, accOut[nt], false, false);
            }
        }

        // ---- Epilogue: out[m][e] = h[m][e] (f32 reload, exact) + corr ----
        int toks[8];
        #pragma unroll
        for (int r = 0; r < 8; ++r) toks[r] = x[tokBase + r + 8 * hi];
        #pragma unroll
        for (int nt = 0; nt < 4; ++nt) {
            #pragma unroll
            for (int r = 0; r < 8; ++r) {
                int m = r + 8 * hi;
                int e = nt * 16 + lm;
                float hval = embed[(long)toks[r] * E + e];
                out[(long)(tokBase + m) * E + e] = hval + accOut[nt][r];
            }
        }
    }
}

extern "C" void kernel_launch(void* const* d_in, const int* in_sizes, int n_in,
                              void* d_out, int out_size, void* d_ws, size_t ws_size,
                              hipStream_t stream) {
    const int*           x     = (const int*)d_in[0];
    const float*         embed = (const float*)d_in[1];
    const float*         W1    = (const float*)d_in[2];
    const float*         W2    = (const float*)d_in[3];
    const unsigned char* memb  = (const unsigned char*)d_in[4];
    float*               out   = (float*)d_out;

    domain_embed_wmma<<<NUM_WG, 256, 0, stream>>>(x, embed, W1, W2, memb, out);
}
